// MultiViewTemporalGNN_76836964926323
// MI455X (gfx1250) — matrix operand order repaired and checked by reading, not compiled, not run on hardware
//
#include <hip/hip_runtime.h>
#include <math.h>

// ---------------- problem constants (match reference) ----------------
#define NN     10000     // nodes
#define DM     128       // model dim
#define EE     80000     // edges
#define KV     4         // views
#define WW     6         // evolution bank depth
#define EVv    20000     // edges per view
#define HH     8         // attention heads
#define HDs    16        // head dim
#define THn    4         // trust heads
#define THDs   32        // trust head dim
#define EDIM   32        // edge msg dim
#define ECH    2500      // trust chunk (EVv % ECH == 0)
#define NCH    (EVv/ECH)
#define SORTN  131072    // next pow2 >= EE
#define EPSc   1e-6f

#define CDIV(a,b) (((a)+(b)-1)/(b))
#define DEV __device__ __forceinline__

// ---------------- WMMA types ----------------
typedef __attribute__((ext_vector_type(16))) __bf16 v16bf;
typedef __attribute__((ext_vector_type(8)))  float  v8f;
union BF16x16 { v16bf v; unsigned u32[8]; unsigned short u[16]; };

DEV unsigned short f2bf(float f) {              // fp32 -> bf16 RNE
  unsigned u = __float_as_uint(f);
  u += 0x7fffu + ((u >> 16) & 1u);
  return (unsigned short)(u >> 16);
}
DEV float gelu_t(float x) {                     // tanh-approx gelu (jax default)
  return 0.5f * x * (1.f + tanhf(0.7978845608028654f * (x + 0.044715f * x * x * x)));
}
DEV float sigm(float x) { return 1.f / (1.f + expf(-x)); }
// order-preserving float<->uint encode for atomicMax on signed floats
DEV unsigned encf(float f) {
  unsigned u = __float_as_uint(f);
  return (u & 0x80000000u) ? ~u : (u | 0x80000000u);
}
DEV float decf(unsigned e) {
  return (e & 0x80000000u) ? __uint_as_float(e ^ 0x80000000u) : __uint_as_float(~e);
}

// one 16x16 WMMA against pre-packed bf16 B (pairs (k,k+1) per dword, row stride nstride)
DEV v8f btile_mma(const BF16x16& a, const unsigned* __restrict__ bp, int nstride, v8f acc) {
  BF16x16 b;
#pragma unroll
  for (int v = 0; v < 8; ++v) b.u32[v] = bp[(long)v * nstride];
  return __builtin_amdgcn_wmma_f32_16x16x32_bf16(false, a.v, false, b.v,
                                                 (short)0, acc, false, false);
}

// ---------------- weight pre-pack: fp32 (K,N) -> bf16 pair dwords (K/2, N) ----------------
__global__ void k_pack_bf16(const float* __restrict__ W, unsigned* out, int K2, int N) {
  int i = blockIdx.x * blockDim.x + threadIdx.x;
  if (i >= K2 * N) return;
  int k2 = i / N, n = i % N;
  unsigned lo = f2bf(W[(long)(2 * k2)     * N + n]);
  unsigned hi = f2bf(W[(long)(2 * k2 + 1) * N + n]);
  out[i] = lo | (hi << 16);
}

// ---------------- generic WMMA GEMM, 16x64 strip per wave ----------------
// C[M,N] = act( (accum? C : 0) + A[rows,K]@B[K,N] + bias ), bf16 ops, fp32 accum.
// A fp32 (optionally row-gathered), staged via LDS; B pre-packed bf16 dword pairs.
// flags: bits[1:0] act (0 none, 1 relu, 2 gelu), bit2 accumulate into C.
__global__ __launch_bounds__(32)
void gemm_wmma_bf16(const float* __restrict__ A, const int* __restrict__ a_rows,
                    int lda, const unsigned* __restrict__ Bp, int nstride,
                    const float* __restrict__ bias, float* __restrict__ C,
                    int ldc, int M, int Ktot, int flags)
{
  __shared__ float sA[16][32];
  const int lane = threadIdx.x;
  const int half = lane >> 4;
  const int lsub = lane & 15;
  const int tm = blockIdx.y << 4;
  const int tn = blockIdx.x << 6;          // 64-wide column strip

  // cooperative A stager: lane loads row (lane/2), 16 cols at ((lane&1)*16)
  const int lrow  = lane >> 1;
  const int lcolb = (lane & 1) << 4;
  int am = tm + lrow; if (am > M - 1) am = M - 1;
  const long aoff = (long)(a_rows ? a_rows[am] : am) * (long)lda;

  v8f acc0 = {}, acc1 = {}, acc2 = {}, acc3 = {};
  for (int kt = 0; kt < Ktot; kt += 32) {
    const float* ap = A + aoff + kt + lcolb;
    if (kt + 32 < Ktot) __builtin_prefetch(ap + 32, 0, 1);
#pragma unroll
    for (int q = 0; q < 4; ++q) {
      float4 f = *(const float4*)(ap + 4 * q);
      sA[lrow][lcolb + 4 * q + 0] = f.x;
      sA[lrow][lcolb + 4 * q + 1] = f.y;
      sA[lrow][lcolb + 4 * q + 2] = f.z;
      sA[lrow][lcolb + 4 * q + 3] = f.w;
    }
    __syncthreads();

    // A 16x32 bf16 lane layout (ISA 7.12.2): lane row = lsub
    BF16x16 a;
#pragma unroll
    for (int v = 0; v < 8; ++v) {
      int k0 = half * 16 + v * 2;
      a.u[2 * v]     = f2bf(sA[lsub][k0]);
      a.u[2 * v + 1] = f2bf(sA[lsub][k0 + 1]);
    }
    // packed B row base: (kt + half*16)/2 ; lanes 0-15 K lo-half, 16-31 K hi-half
    const int kb2 = (kt >> 1) + (half << 3);
    const unsigned* bp = Bp + (long)kb2 * nstride + tn + lsub;
    acc0 = btile_mma(a, bp +  0, nstride, acc0);
    acc1 = btile_mma(a, bp + 16, nstride, acc1);
    acc2 = btile_mma(a, bp + 32, nstride, acc2);
    acc3 = btile_mma(a, bp + 48, nstride, acc3);
    __syncthreads();
  }

  const int act = flags & 3;
  const bool accum = (flags & 4) != 0;
  v8f accs[4] = {acc0, acc1, acc2, acc3};
#pragma unroll
  for (int t = 0; t < 4; ++t) {
    const int gn = tn + t * 16 + lsub;
#pragma unroll
    for (int v = 0; v < 8; ++v) {
      int gm = tm + v + half * 8;          // D layout: VGPR v -> row v + 8*half
      if (gm >= M) continue;
      float val = accs[t][v];
      if (accum) val += C[(long)gm * ldc + gn];
      if (bias)  val += bias[gn];
      if (act == 1)      val = fmaxf(val, 0.f);
      else if (act == 2) val = gelu_t(val);
      C[(long)gm * ldc + gn] = val;
    }
  }
}

// ---------------- sort (argsort of edge_t via bitonic) ----------------
__global__ void k_sort_init(const float* __restrict__ t, float* keys, int* idx) {
  int i = blockIdx.x * blockDim.x + threadIdx.x;
  if (i >= SORTN) return;
  keys[i] = (i < EE) ? t[i] : 3.4e38f;
  idx[i] = i;
}
__global__ void k_bitonic(float* keys, int* idx, int j, int k) {
  int i = blockIdx.x * blockDim.x + threadIdx.x;
  if (i >= SORTN) return;
  int p = i ^ j;
  if (p > i) {
    float a = keys[i], b = keys[p];
    bool up = ((i & k) == 0);
    if (up ? (a > b) : (a < b)) {
      keys[i] = b; keys[p] = a;
      int ti = idx[i]; idx[i] = idx[p]; idx[p] = ti;
    }
  }
}

// ---------------- small utility / graph kernels ----------------
__global__ void k_fill_u32(unsigned* p, unsigned v, long n) {
  long i = (long)blockIdx.x * blockDim.x + threadIdx.x;
  if (i < n) p[i] = v;
}
__global__ void k_conf(const float* __restrict__ mem_v, float* conf) {
  int n = blockIdx.x * blockDim.x + threadIdx.x;
  if (n >= NN) return;
  float s = 0.f;
  for (int d = 0; d < DM; ++d) s += mem_v[(long)n * DM + d];
  float c = fmaxf(sqrtf(s / DM), EPSc);
  conf[n] = 1.f / (1.f + c);
}
__global__ void k_gather_view(const int* __restrict__ ei, const float* __restrict__ et,
                              const int* __restrict__ order, int kview,
                              int* vsrc, int* vdst, float* vt) {
  int e = blockIdx.x * blockDim.x + threadIdx.x;
  if (e >= EVv) return;
  int ge = order[kview * EVv + e];
  vsrc[e] = ei[ge];
  vdst[e] = ei[EE + ge];
  vt[e]   = et[ge];
}
// edges in a view are time-sorted; per-source scan gives inter-event deltas
__global__ void k_delta(const int* __restrict__ vsrc, const float* __restrict__ vt,
                        float* delta) {
  int n = blockIdx.x * blockDim.x + threadIdx.x;
  if (n >= NN) return;
  float prev = 0.f; bool seen = false;
  for (int e = 0; e < EVv; ++e) {
    if (vsrc[e] == n) {
      delta[e] = seen ? (vt[e] - prev) : 0.f;
      prev = vt[e]; seen = true;
    }
  }
}
__global__ void k_time_embed(const float* __restrict__ vt, const float* __restrict__ delta,
                             const float* __restrict__ Wt, const float* __restrict__ bt,
                             float* temb) {
  long i = (long)blockIdx.x * blockDim.x + threadIdx.x;
  if (i >= (long)EVv * DM) return;
  int e = (int)(i >> 7), d = (int)(i & 127);
  float z = vt[e] * Wt[d] + delta[e] * Wt[DM + d] + bt[d];
  temb[i] = (d == 0) ? z : sinf(z);
}
__global__ void k_scores(const float* __restrict__ Qn, const float* __restrict__ Kn,
                         const float* __restrict__ eK, const int* __restrict__ vsrc,
                         const int* __restrict__ vdst, float* scores, unsigned* nmax) {
  int e = blockIdx.x * blockDim.x + threadIdx.x;
  if (e >= EVv) return;
  int s = vsrc[e], d = vdst[e];
  const float* q  = Qn + (long)d * DM;
  const float* kk = Kn + (long)s * DM;
  const float* ek = eK + (long)e * DM;
  for (int h = 0; h < HH; ++h) {
    float acc = 0.f;
    for (int j = 0; j < HDs; ++j) { int c = h * HDs + j; acc += q[c] * (kk[c] + ek[c]); }
    acc *= 0.25f;                       // HD^-0.5
    scores[(long)e * HH + h] = acc;
    atomicMax(&nmax[(long)d * HH + h], encf(acc));
  }
}
__global__ void k_expsum(const float* __restrict__ scores, const unsigned* __restrict__ nmax,
                         const int* __restrict__ vdst, float* ex, float* nsum) {
  long i = (long)blockIdx.x * blockDim.x + threadIdx.x;
  if (i >= (long)EVv * HH) return;
  int e = (int)(i / HH), h = (int)(i % HH);
  int d = vdst[e];
  float m = decf(nmax[(long)d * HH + h]);
  if (!isfinite(m)) m = 0.f;
  float v = expf(scores[i] - m);
  ex[i] = v;
  atomicAdd(&nsum[(long)d * HH + h], v);
}
__global__ void k_rowmap(const int* __restrict__ vsrc, int e0, int ec, int* rowmap) {
  int r = blockIdx.x * blockDim.x + threadIdx.x;
  if (r >= ec * WW) return;
  rowmap[r] = vsrc[e0 + r / WW] * WW + (r % WW);
}
__global__ void k_trust_attn(const float* __restrict__ Qt, const float* __restrict__ Kt,
                             const float* __restrict__ Vt, const float* __restrict__ logdec,
                             float* pooled, int ec) {
  int idx = blockIdx.x * blockDim.x + threadIdx.x;
  if (idx >= ec * THn) return;
  int e = idx / THn, h = idx % THn;
  const float* q = Qt + (long)e * DM + h * THDs;
  float dw[WW], dsum = 0.f, rr = expf(logdec[0]);
  for (int w = 0; w < WW; ++w) { dw[w] = expf(rr * (float)w); dsum += dw[w]; }
  float sc[WW], mx = -1e30f;
  for (int w = 0; w < WW; ++w) {
    const float* kr = Kt + ((long)e * WW + w) * DM + h * THDs;
    float a = 0.f;
    for (int j = 0; j < THDs; ++j) a += q[j] * kr[j];
    a = a * 0.17677669529663687f + dw[w] / dsum;      // THD^-0.5
    sc[w] = a; mx = fmaxf(mx, a);
  }
  float se = 0.f;
  for (int w = 0; w < WW; ++w) { sc[w] = expf(sc[w] - mx); se += sc[w]; }
  float inv = 1.f / se;
  for (int j = 0; j < THDs; ++j) {
    float p = 0.f;
    for (int w = 0; w < WW; ++w) p += sc[w] * Vt[((long)e * WW + w) * DM + h * THDs + j];
    pooled[(long)e * DM + h * THDs + j] = p * inv;
  }
}
__global__ void k_out2(const float* __restrict__ h1t, const float* __restrict__ W2,
                       const float* __restrict__ b2, const float* __restrict__ conf,
                       const int* __restrict__ vsrc, const int* __restrict__ vdst,
                       int e0, int ec, float* trustg) {
  int e = blockIdx.x * blockDim.x + threadIdx.x;
  if (e >= ec) return;
  float a = b2[0];
  for (int j = 0; j < 64; ++j) a += h1t[(long)e * 64 + j] * W2[j];
  int ge = e0 + e;
  float lc = 0.5f * (logf(conf[vdst[ge]]) + logf(conf[vsrc[ge]]));
  lc = fmaxf(lc, -5.f);
  trustg[ge] = sigm(a + lc);
}
__global__ void k_gate(const float* __restrict__ x, const int* __restrict__ vsrc,
                       const int* __restrict__ vdst, const float* __restrict__ trustg,
                       const float* __restrict__ gate_bal, float* g) {
  int e = blockIdx.x * blockDim.x + threadIdx.x;
  if (e >= EVv) return;
  const float* xs = x + (long)vsrc[e] * DM;
  const float* xd = x + (long)vdst[e] * DM;
  float dot = 0.f, na = 0.f, nb = 0.f;
  for (int d = 0; d < DM; ++d) { float a = xs[d], b = xd[d]; dot += a * b; na += a * a; nb += b * b; }
  float cos = dot / (sqrtf(na) * sqrtf(nb) + EPSc);
  float sim = sigm(cos * 2.0f);                      // 1/SIM_TEMP
  float gb = sigm(gate_bal[0]);
  g[e] = gb * trustg[e] + (1.f - gb) * sim;
}
__global__ void k_aggregate(const float* __restrict__ ex, const float* __restrict__ nsum,
                            const float* __restrict__ g, const float* __restrict__ Vn,
                            const int* __restrict__ vsrc, const int* __restrict__ vdst,
                            float* agg) {
  long i = (long)blockIdx.x * blockDim.x + threadIdx.x;
  if (i >= (long)EVv * DM) return;
  int e = (int)(i >> 7), dcol = (int)(i & 127), h = dcol >> 4;
  int s = vsrc[e], d = vdst[e];
  float alpha = ex[(long)e * HH + h] / (nsum[(long)d * HH + h] + EPSc);
  atomicAdd(&agg[(long)d * DM + dcol], alpha * g[e] * Vn[(long)s * DM + dcol]);
}
__global__ void k_rgate(const float* __restrict__ x, const float* __restrict__ Wg,
                        const float* __restrict__ bg, float* r) {
  int n = blockIdx.x * blockDim.x + threadIdx.x;
  if (n >= NN) return;
  float a = bg[0];
  for (int d = 0; d < DM; ++d) a += x[(long)n * DM + d] * Wg[d];
  r[n] = sigm(a);
}
__global__ void k_update_ln(const float* __restrict__ x, const float* __restrict__ rg,
                            const float* __restrict__ fused, const float* __restrict__ lng,
                            const float* __restrict__ lnb, float* xnew, float* out, int kview) {
  int n = blockIdx.x * blockDim.x + threadIdx.x;
  if (n >= NN) return;
  float rv = rg[n], s = 0.f, ss = 0.f;
  const float* xr = x + (long)n * DM;
  const float* fr = fused + (long)n * DM;
  float* xo = xnew + (long)n * DM;
  float* orow = out + ((long)n * KV + kview) * DM;
  for (int d = 0; d < DM; ++d) { float h = xr[d] + rv * fr[d]; xo[d] = h; s += h; ss += h * h; }
  float mu = s / DM;
  float var = fmaxf(ss / DM - mu * mu, 0.f);
  float inv = rsqrtf(var + 1e-5f);
  for (int d = 0; d < DM; ++d) {
    float v = (xo[d] - mu) * inv * lng[d] + lnb[d];
    xo[d] = v; orow[d] = v;
  }
}

// ---------------- host orchestration ----------------
static inline char* carve(char* base, size_t& off, size_t bytes) {
  char* p = base + off;
  off = (off + bytes + 255) & ~(size_t)255;
  return p;
}
static inline void gemm(hipStream_t s, const float* A, const int* rows, int lda,
                        const unsigned* Bp, int nstride, const float* bias, float* C,
                        int ldc, int M, int K, int Ncols, int flags) {
  dim3 grid(Ncols / 64, CDIV(M, 16));
  gemm_wmma_bf16<<<grid, dim3(32), 0, s>>>(A, rows, lda, Bp, nstride, bias, C, ldc, M, K, flags);
}
static inline void packw(hipStream_t s, const float* W, unsigned* out, int K, int N) {
  int n = (K / 2) * N;
  k_pack_bf16<<<CDIV(n, 256), 256, 0, s>>>(W, out, K / 2, N);
}
enum { ACT_NONE = 0, ACT_RELU = 1, ACT_GELU = 2, ACCUM = 4 };

extern "C" void kernel_launch(void* const* d_in, const int* in_sizes, int n_in,
                              void* d_out, int out_size, void* d_ws, size_t ws_size,
                              hipStream_t stream) {
  (void)in_sizes; (void)n_in; (void)out_size; (void)ws_size;
  // setup_inputs() dict order; params flattened pytree-style (sorted keys, (W,b) tuples)
  const float* x_in    = (const float*)d_in[0];
  const int*   eidx    = (const int*)  d_in[1];
  const float* et      = (const float*)d_in[2];
  const float* emsg    = (const float*)d_in[3];
  const float* We_enc  = (const float*)d_in[4];
  const float* be_enc  = (const float*)d_in[5];
  const float* evo     = (const float*)d_in[6];
  const float* W_edge  = (const float*)d_in[7];
  const float* b_edge  = (const float*)d_in[8];
  const float* W_f1    = (const float*)d_in[9];
  const float* b_f1    = (const float*)d_in[10];
  const float* W_f2    = (const float*)d_in[11];
  const float* b_f2    = (const float*)d_in[12];
  const float* W_gate  = (const float*)d_in[13];
  const float* b_gate  = (const float*)d_in[14];
  const float* gatebal = (const float*)d_in[15];
  const float* W_k     = (const float*)d_in[16];
  const float* b_k     = (const float*)d_in[17];
  const float* ln_b    = (const float*)d_in[18];
  const float* ln_g    = (const float*)d_in[19];
  const float* W_o     = (const float*)d_in[20];
  const float* b_o     = (const float*)d_in[21];
  const float* W_q     = (const float*)d_in[22];
  const float* b_q     = (const float*)d_in[23];
  const float* W_t     = (const float*)d_in[24];
  const float* b_t     = (const float*)d_in[25];
  const float* Wt_k    = (const float*)d_in[26];
  const float* bt_k    = (const float*)d_in[27];
  const float* logdec  = (const float*)d_in[28];
  const float* Wt_o1   = (const float*)d_in[29];
  const float* bt_o1   = (const float*)d_in[30];
  const float* Wt_o2   = (const float*)d_in[31];
  const float* bt_o2   = (const float*)d_in[32];
  const float* Wt_q    = (const float*)d_in[33];
  const float* bt_q    = (const float*)d_in[34];
  const float* Wt_v    = (const float*)d_in[35];
  const float* bt_v    = (const float*)d_in[36];
  const float* W_v     = (const float*)d_in[37];
  const float* b_v     = (const float*)d_in[38];
  const float* mem_h   = (const float*)d_in[39];
  const float* mem_v   = (const float*)d_in[40];
  const float* Wtime2  = (const float*)d_in[41];
  const float* btime2  = (const float*)d_in[42];
  float* out = (float*)d_out;

  // workspace carve (~100 MB)
  char* base = (char*)d_ws; size_t off = 0;
  float*    skeys  = (float*)   carve(base, off, (size_t)SORTN * 4);
  int*      sidx   = (int*)     carve(base, off, (size_t)SORTN * 4);
  float*    xa     = (float*)   carve(base, off, (size_t)NN * DM * 4);
  float*    xb     = (float*)   carve(base, off, (size_t)NN * DM * 4);
  int*      vsrc   = (int*)     carve(base, off, (size_t)EVv * 4);
  int*      vdst   = (int*)     carve(base, off, (size_t)EVv * 4);
  float*    vt     = (float*)   carve(base, off, (size_t)EVv * 4);
  float*    vdel   = (float*)   carve(base, off, (size_t)EVv * 4);
  float*    eemb   = (float*)   carve(base, off, (size_t)EVv * DM * 4);
  float*    temb   = (float*)   carve(base, off, (size_t)EVv * DM * 4);
  float*    eK     = (float*)   carve(base, off, (size_t)EVv * DM * 4);
  float*    Qn     = (float*)   carve(base, off, (size_t)NN * DM * 4);
  float*    Kn     = (float*)   carve(base, off, (size_t)NN * DM * 4);
  float*    Vn     = (float*)   carve(base, off, (size_t)NN * DM * 4);
  float*    scores = (float*)   carve(base, off, (size_t)EVv * HH * 4);
  float*    exb    = (float*)   carve(base, off, (size_t)EVv * HH * 4);
  unsigned* nmax   = (unsigned*)carve(base, off, (size_t)NN * HH * 4);
  float*    nsum   = (float*)   carve(base, off, (size_t)NN * HH * 4);
  float*    confb  = (float*)   carve(base, off, (size_t)NN * 4);
  float*    trustg = (float*)   carve(base, off, (size_t)EVv * 4);
  float*    gb     = (float*)   carve(base, off, (size_t)EVv * 4);
  int*      rowmap = (int*)     carve(base, off, (size_t)ECH * WW * 4);
  float*    Qt     = (float*)   carve(base, off, (size_t)ECH * DM * 4);
  float*    Kt     = (float*)   carve(base, off, (size_t)ECH * WW * DM * 4);
  float*    Vt     = (float*)   carve(base, off, (size_t)ECH * WW * DM * 4);
  float*    pooled = (float*)   carve(base, off, (size_t)ECH * DM * 4);
  float*    h1t    = (float*)   carve(base, off, (size_t)ECH * 64 * 4);
  float*    agg    = (float*)   carve(base, off, (size_t)NN * DM * 4);
  float*    obuf   = (float*)   carve(base, off, (size_t)NN * DM * 4);
  float*    h1     = (float*)   carve(base, off, (size_t)NN * DM * 4);
  float*    fusedb = (float*)   carve(base, off, (size_t)NN * DM * 4);
  float*    rg     = (float*)   carve(base, off, (size_t)NN * 4);
  // packed bf16 weights (dword = bf16 pair (k,k+1)); L2-resident, reused many times
  unsigned* pq   = (unsigned*)carve(base, off, 8192 * 4);
  unsigned* pk   = (unsigned*)carve(base, off, 8192 * 4);
  unsigned* pv   = (unsigned*)carve(base, off, 8192 * 4);
  unsigned* po   = (unsigned*)carve(base, off, 8192 * 4);
  unsigned* pedg = (unsigned*)carve(base, off, 8192 * 4);
  unsigned* ptim = (unsigned*)carve(base, off, 8192 * 4);
  unsigned* pf1  = (unsigned*)carve(base, off, 16384 * 4);
  unsigned* pf2  = (unsigned*)carve(base, off, 8192 * 4);
  unsigned* ptq  = (unsigned*)carve(base, off, 8192 * 4);
  unsigned* ptk  = (unsigned*)carve(base, off, 8192 * 4);
  unsigned* ptv  = (unsigned*)carve(base, off, 8192 * 4);
  unsigned* po1  = (unsigned*)carve(base, off, 4096 * 4);
  unsigned* penc = (unsigned*)carve(base, off, 2048 * 4);

  hipMemcpyAsync(xa, x_in, (size_t)NN * DM * 4, hipMemcpyDeviceToDevice, stream);

  // pack all GEMM weights to bf16 once per launch
  packw(stream, W_q, pq, DM, DM);      packw(stream, W_k, pk, DM, DM);
  packw(stream, W_v, pv, DM, DM);      packw(stream, W_o, po, DM, DM);
  packw(stream, W_edge, pedg, DM, DM); packw(stream, W_t, ptim, DM, DM);
  packw(stream, W_f1, pf1, 2 * DM, DM);
  packw(stream, W_f2, pf2, DM, DM);
  packw(stream, Wt_q, ptq, DM, DM);    packw(stream, Wt_k, ptk, DM, DM);
  packw(stream, Wt_v, ptv, DM, DM);    packw(stream, Wt_o1, po1, DM, 64);
  packw(stream, We_enc, penc, EDIM, DM);

  // argsort(edge_t): bitonic over padded pow2
  k_sort_init<<<SORTN / 256, 256, 0, stream>>>(et, skeys, sidx);
  for (int k = 2; k <= SORTN; k <<= 1)
    for (int j = k >> 1; j > 0; j >>= 1)
      k_bitonic<<<SORTN / 256, 256, 0, stream>>>(skeys, sidx, j, k);

  k_conf<<<CDIV(NN, 256), 256, 0, stream>>>(mem_v, confb);

  float* cur = xa; float* nxt = xb;
  for (int kv = 0; kv < KV; ++kv) {
    const int* order = sidx;                 // first EE sorted entries are real edges
    k_gather_view<<<CDIV(EVv, 256), 256, 0, stream>>>(eidx, et, order, kv, vsrc, vdst, vt);
    k_delta<<<CDIV(NN, 256), 256, 0, stream>>>(vsrc, vt, vdel);

    // node projections on current x
    gemm(stream, cur, nullptr, DM, pq, DM, b_q, Qn, DM, NN, DM, DM, ACT_NONE);
    gemm(stream, cur, nullptr, DM, pk, DM, b_k, Kn, DM, NN, DM, DM, ACT_NONE);
    gemm(stream, cur, nullptr, DM, pv, DM, b_v, Vn, DM, NN, DM, DM, ACT_NONE);
    k_rgate<<<CDIV(NN, 256), 256, 0, stream>>>(cur, W_gate, b_gate, rg);

    // edge_emb = relu(msg[view] @ We + be)   (A gathered by view edge ids)
    gemm(stream, emsg, order + kv * EVv, EDIM, penc, DM, be_enc, eemb, DM, EVv, EDIM, DM, ACT_RELU);
    k_time_embed<<<CDIV((long)EVv * DM, 256), 256, 0, stream>>>(vt, vdel, Wtime2, btime2, temb);

    // eK = edge_emb@W_edge + b_edge + time_emb@W_time + b_time  (accumulating GEMM pair)
    gemm(stream, eemb, nullptr, DM, pedg, DM, b_edge, eK, DM, EVv, DM, DM, ACT_NONE);
    gemm(stream, temb, nullptr, DM, ptim, DM, b_t,    eK, DM, EVv, DM, DM, ACCUM | ACT_NONE);

    // segment softmax over dst
    k_fill_u32<<<CDIV((long)NN * HH, 256), 256, 0, stream>>>(nmax, 0u, (long)NN * HH);
    k_fill_u32<<<CDIV((long)NN * HH, 256), 256, 0, stream>>>((unsigned*)nsum, 0u, (long)NN * HH);
    k_scores<<<CDIV(EVv, 256), 256, 0, stream>>>(Qn, Kn, eK, vsrc, vdst, scores, nmax);
    k_expsum<<<CDIV((long)EVv * HH, 256), 256, 0, stream>>>(scores, nmax, vdst, exb, nsum);

    // trust module, chunked so K/V activations stay small
    for (int c = 0; c < NCH; ++c) {
      int e0 = c * ECH;
      k_rowmap<<<CDIV(ECH * WW, 256), 256, 0, stream>>>(vsrc, e0, ECH, rowmap);
      gemm(stream, mem_h, vdst + e0, DM, ptq, DM, bt_q, Qt, DM, ECH, DM, DM, ACT_NONE);
      gemm(stream, evo, rowmap, DM, ptk, DM, bt_k, Kt, DM, ECH * WW, DM, DM, ACT_NONE);
      gemm(stream, evo, rowmap, DM, ptv, DM, bt_v, Vt, DM, ECH * WW, DM, DM, ACT_NONE);
      k_trust_attn<<<CDIV(ECH * THn, 256), 256, 0, stream>>>(Qt, Kt, Vt, logdec, pooled, ECH);
      gemm(stream, pooled, nullptr, DM, po1, 64, bt_o1, h1t, 64, ECH, DM, 64, ACT_GELU);
      k_out2<<<CDIV(ECH, 256), 256, 0, stream>>>(h1t, Wt_o2, bt_o2, confb, vsrc, vdst, e0, ECH, trustg);
    }

    // combined edge gate, aggregate messages
    k_gate<<<CDIV(EVv, 256), 256, 0, stream>>>(cur, vsrc, vdst, trustg, gatebal, gb);
    k_fill_u32<<<CDIV((long)NN * DM, 256), 256, 0, stream>>>((unsigned*)agg, 0u, (long)NN * DM);
    k_aggregate<<<CDIV((long)EVv * DM, 256), 256, 0, stream>>>(exb, nsum, gb, Vn, vsrc, vdst, agg);

    // output proj + fused MLP (f1 over concat(x,o) as two accumulating GEMMs)
    gemm(stream, agg, nullptr, DM, po, DM, b_o, obuf, DM, NN, DM, DM, ACT_NONE);
    gemm(stream, cur,  nullptr, DM, pf1,            DM, nullptr, h1, DM, NN, DM, DM, ACT_NONE);
    gemm(stream, obuf, nullptr, DM, pf1 + 64 * DM,  DM, b_f1,    h1, DM, NN, DM, DM, ACCUM | ACT_GELU);
    gemm(stream, h1, nullptr, DM, pf2, DM, b_f2, fusedb, DM, NN, DM, DM, ACT_NONE);

    // gated residual + layernorm; emit view slice
    k_update_ln<<<CDIV(NN, 256), 256, 0, stream>>>(cur, rg, fusedb, ln_g, ln_b, nxt, out, kv);
    float* tmp = cur; cur = nxt; nxt = tmp;
  }
}
